// ComplexDifferentialAttention_68410239090889
// MI455X (gfx1250) — matrix operand order repaired and checked by reading, not compiled, not run on hardware
//
#include <hip/hip_runtime.h>
#include <hip/hip_bf16.h>
#include <math.h>

// ---------------------------------------------------------------------------
// ComplexDifferentialAttention on gfx1250 (MI455X).
// Matmuls: v_wmma_f32_16x16x32_f16 (f16 in, f32 accumulate).
// K tiles in attention: double-buffered global_load_async_to_lds_b128.
// ---------------------------------------------------------------------------

typedef __attribute__((ext_vector_type(16))) _Float16 v16h;
typedef __attribute__((ext_vector_type(8)))  float    v8f;

constexpr int   kH    = 12;
constexpr int   kS    = 1024;
constexpr int   kD    = 128;
constexpr int   kHS   = kH * kS;        // 12288 tokens
constexpr int   kHSD  = kHS * kD;       // 1572864 elements
constexpr float kScale      = 0.08838834764831845f;   // D^-0.5
constexpr float kLambdaInit = 0.35550906759096927f;   // 0.8 - 0.6*exp(-0.3)

// Attention LDS map (dynamic): [0, 128K) P/mag rows (32 x 4096B),
// [128K, 128K+16K) double-buffered K tile stage (8KB per buffer: r then i).
constexpr unsigned kPBytes     = 32u * 4096u;   // 131072
constexpr unsigned kStageBytes = 2u * 8192u;    // 16384
constexpr unsigned kAttnLds    = kPBytes + kStageBytes;

__device__ __forceinline__ v8f wmma32(v16h a, v16h b, v8f c) {
  return __builtin_amdgcn_wmma_f32_16x16x32_f16(
      /*neg_a=*/false, a, /*neg_b=*/false, b,
      /*c_mod=*/(short)0, c, /*reuse_a=*/false, /*reuse_b=*/false);
}

// ---- Async-to-LDS helpers (CDNA5 ASYNCcnt path) ---------------------------
__device__ __forceinline__ void async_ld_b128(unsigned lds_off,
                                              const void* gaddr) {
  asm volatile("global_load_async_to_lds_b128 %0, %1, off"
               :: "v"(lds_off), "v"(gaddr) : "memory");
}
__device__ __forceinline__ void wait_async_le8() {
  asm volatile("s_wait_asynccnt 0x8" ::: "memory");
}
__device__ __forceinline__ void wait_async_0() {
  asm volatile("s_wait_asynccnt 0x0" ::: "memory");
}

// ---- Fragment loaders (layouts per CDNA5 ISA 7.12.2) ----------------------
// A (16x32, f16): lane=(r,hi); row M=r; elems 0..7 -> K=hi*8+j,
//                 elems 8..15 -> K=16+hi*8+j.
// B (32x16, f16): lane=(r,hi); col N=r; elems 0..7 -> K=hi*8+j,
//                 elems 8..15 -> K=16+hi*8+j.
// C (16x16, f32): elem j -> row M=j+8*hi, col N=r.

__device__ __forceinline__ v16h load_a_f32(const float* X, int ld, int row0,
                                           int k0, int lane) {
  int r = lane & 15, hi = lane >> 4;
  const float* p = X + (size_t)(row0 + r) * ld + k0;
  v16h a;
#pragma unroll
  for (int j = 0; j < 8; ++j) {
    a[j]     = (_Float16)p[hi * 8 + j];
    a[8 + j] = (_Float16)p[16 + hi * 8 + j];
  }
  return a;
}

__device__ __forceinline__ v16h load_bT_f32(const float* W, int ld, int col0,
                                            int k0, int lane) {
  int r = lane & 15, hi = lane >> 4;
  const float* p = W + (size_t)(col0 + r) * ld + k0;
  v16h b;
#pragma unroll
  for (int j = 0; j < 8; ++j) {
    b[j]     = (_Float16)p[hi * 8 + j];
    b[8 + j] = (_Float16)p[16 + hi * 8 + j];
  }
  return b;
}

__device__ __forceinline__ v16h load_a_f16(const _Float16* X, int ld, int row0,
                                           int k0, int lane) {
  int r = lane & 15, hi = lane >> 4;
  const _Float16* p = X + (size_t)(row0 + r) * ld + k0;
  v16h a;
#pragma unroll
  for (int j = 0; j < 8; ++j) {
    a[j]     = p[hi * 8 + j];
    a[8 + j] = p[16 + hi * 8 + j];
  }
  return a;
}

__device__ __forceinline__ v16h load_bT_f16(const _Float16* W, int ld, int col0,
                                            int k0, int lane) {
  int r = lane & 15, hi = lane >> 4;
  const _Float16* p = W + (size_t)(col0 + r) * ld + k0;
  v16h b;
#pragma unroll
  for (int j = 0; j < 8; ++j) {
    b[j]     = p[hi * 8 + j];
    b[8 + j] = p[16 + hi * 8 + j];
  }
  return b;
}

// ---------------------------------------------------------------------------
// Complex projection:  o_r = xr*wr^T - xi*wi^T + br (+pe)
//                      o_i = xr*wi^T + xi*wr^T + bi (+pe)
// x: [kHS, kD] f32, w: [O, kD] f32.  One wave per 16x16 tile, K=128.
// ---------------------------------------------------------------------------
enum { PROJ_ROW16 = 0, PROJ_TR16 = 1, PROJ_F32 = 2 };

__global__ void cda_proj_kernel(const float* __restrict__ xr,
                                const float* __restrict__ xi,
                                const float* __restrict__ wr,
                                const float* __restrict__ wi,
                                const float* __restrict__ br,
                                const float* __restrict__ bi,
                                const float* __restrict__ per,
                                const float* __restrict__ pei,
                                int O, int mode,
                                _Float16* __restrict__ o16r,
                                _Float16* __restrict__ o16i,
                                float* __restrict__ o32r,
                                float* __restrict__ o32i) {
  const int row0 = blockIdx.x * 16;
  const int col0 = blockIdx.y * 16;
  const int lane = threadIdx.x & 31;

  v8f acc_rr = {}, acc_ii = {}, acc_ri = {}, acc_ir = {};
#pragma unroll
  for (int kk = 0; kk < 4; ++kk) {
    const int k0 = kk * 32;
    v16h axr = load_a_f32(xr, kD, row0, k0, lane);
    v16h axi = load_a_f32(xi, kD, row0, k0, lane);
    v16h bwr = load_bT_f32(wr, kD, col0, k0, lane);
    v16h bwi = load_bT_f32(wi, kD, col0, k0, lane);
    acc_rr = wmma32(axr, bwr, acc_rr);  // xr*wr
    acc_ii = wmma32(axi, bwi, acc_ii);  // xi*wi
    acc_ri = wmma32(axr, bwi, acc_ri);  // xr*wi
    acc_ir = wmma32(axi, bwr, acc_ir);  // xi*wr
  }

  const int r = lane & 15, hi = lane >> 4;
#pragma unroll
  for (int j = 0; j < 8; ++j) {
    const int row = row0 + j + 8 * hi;
    const int col = col0 + r;
    float o_r = acc_rr[j] - acc_ii[j] + br[col];
    float o_i = acc_ri[j] + acc_ir[j] + bi[col];
    if (per != nullptr) {  // PE add (q: duplicated across the 2D halves)
      o_r += per[(size_t)row * kD + (col & (kD - 1))];
      o_i += pei[(size_t)row * kD + (col & (kD - 1))];
    }
    if (mode == PROJ_ROW16) {
      o16r[(size_t)row * O + col] = (_Float16)o_r;
      o16i[(size_t)row * O + col] = (_Float16)o_i;
    } else if (mode == PROJ_TR16) {
      const int h = row >> 10, s = row & (kS - 1);
      const size_t idx = (size_t)h * kD * kS + (size_t)col * kS + s;
      o16r[idx] = (_Float16)o_r;
      o16i[idx] = (_Float16)o_i;
    } else {
      o32r[(size_t)row * O + col] = o_r;
      o32i[(size_t)row * O + col] = o_i;
    }
  }
}

// ---------------------------------------------------------------------------
// Attention: one workgroup (2 waves) per (head, 32-query block, split).
// Phase1: scores via WMMA with K tiles double-buffer async-staged in LDS.
// Phase2: softmax; P kept as unnormalized exp() in f16 (in place),
//         1/sum applied in the AV epilogue.
// Phase3: P @ V via WMMA.
// ---------------------------------------------------------------------------
__global__ void cda_attn_kernel(const _Float16* __restrict__ qp_r,
                                const _Float16* __restrict__ qp_i,
                                const _Float16* __restrict__ kp_r,
                                const _Float16* __restrict__ kp_i,
                                const _Float16* __restrict__ vt_r,
                                const _Float16* __restrict__ vt_i,
                                float* __restrict__ a_r,
                                float* __restrict__ a_i) {
  extern __shared__ __align__(16) char lds[];  // kAttnLds bytes
  __shared__ float smax2[64];
  __shared__ float sinv[32];

  const int qblk  = blockIdx.x;
  const int h     = blockIdx.y;
  const int split = blockIdx.z;
  const int tid   = threadIdx.x;     // 0..63
  const int w     = tid >> 5;
  const int lane  = tid & 31;
  const int r = lane & 15, hi = lane >> 4;

  const int qrow0 = h * kS + qblk * 32 + w * 16;
  const int qcol0 = split * kD;

  const unsigned stage_base = (unsigned)(size_t)(lds + kPBytes);

  // Issue the 8 async b128 loads (per thread's share) for K tile kt -> buf.
  auto issue_tile = [&](int kt, int buf) {
    const int krow0 = h * kS + kt * 16;
    const char* gr = (const char*)(kp_r + (size_t)krow0 * kD);
    const char* gi = (const char*)(kp_i + (size_t)krow0 * kD);
    const unsigned lb = stage_base + (unsigned)buf * 8192u;
#pragma unroll
    for (int it = 0; it < 4; ++it) {
      const int c = it * 64 + tid;        // 16B chunk index within 4KB
      async_ld_b128(lb + c * 16, gr + c * 16);
      async_ld_b128(lb + 4096u + c * 16, gi + c * 16);
    }
  };

  // ---- Phase 1: score magnitudes for 16 queries x 1024 keys ----
  issue_tile(0, 0);
  for (int kt = 0; kt < kS / 16; ++kt) {
    const int buf = kt & 1;
    if (kt + 1 < kS / 16) {
      issue_tile(kt + 1, buf ^ 1);
      wait_async_le8();   // 8 newest in flight; tile kt's loads complete
    } else {
      wait_async_0();
    }
    __syncthreads();      // staged tile visible to both waves

    const _Float16* kbr = (const _Float16*)(lds + kPBytes + buf * 8192);
    const _Float16* kbi = (const _Float16*)(lds + kPBytes + buf * 8192 + 4096);

    v8f arr = {}, aii = {}, air = {}, ari = {};
#pragma unroll
    for (int kk = 0; kk < 4; ++kk) {
      const int k0 = kk * 32;
      v16h aqr = load_a_f16(qp_r, 2 * kD, qrow0, qcol0 + k0, lane);
      v16h aqi = load_a_f16(qp_i, 2 * kD, qrow0, qcol0 + k0, lane);
      v16h bkr = load_bT_f16(kbr, kD, 0, k0, lane);
      v16h bki = load_bT_f16(kbi, kD, 0, k0, lane);
      arr = wmma32(aqr, bkr, arr);
      aii = wmma32(aqi, bki, aii);
      air = wmma32(aqi, bkr, air);
      ari = wmma32(aqr, bki, ari);
    }
#pragma unroll
    for (int j = 0; j < 8; ++j) {
      const float sr  = arr[j] + aii[j];
      const float si  = air[j] - ari[j];
      const float mag = sqrtf(sr * sr + si * si + 1e-8f) * kScale;
      const int rowL  = w * 16 + j + 8 * hi;
      *((float*)(lds + (size_t)rowL * 4096) + (kt * 16 + r)) = mag;
    }
  }
  __syncthreads();

  // ---- Phase 2: softmax. Max parallel 2 threads/row; one fused pass does
  // exp + sum + in-place f16 store (unnormalized; write 2k < read 4k: safe).
  {
    const int row = tid >> 1, half = tid & 1;
    float* m = (float*)(lds + (size_t)row * 4096);
    float mx = -1e30f;
    for (int k = half * 512; k < half * 512 + 512; ++k) mx = fmaxf(mx, m[k]);
    smax2[tid] = mx;
    __syncthreads();
    if (half == 0) {
      const float mxr = fmaxf(smax2[2 * row], smax2[2 * row + 1]);
      _Float16* ph = (_Float16*)m;
      float sum = 0.0f;
      for (int k = 0; k < kS; ++k) {
        const float e = __expf(m[k] - mxr);
        sum += e;
        ph[k] = (_Float16)e;
      }
      sinv[row] = 1.0f / sum;
    }
  }
  __syncthreads();

  // ---- Phase 3: A = (P @ V) * (1/sum)  (r and i) ----
  const _Float16* vtrh = vt_r + (size_t)h * kD * kS;  // [d][s]
  const _Float16* vtih = vt_i + (size_t)h * kD * kS;
  for (int dt = 0; dt < kD / 16; ++dt) {
    v8f accr = {}, acci = {};
    for (int k0 = 0; k0 < kS; k0 += 32) {
      const _Float16* prow =
          (const _Float16*)(lds + (size_t)(w * 16 + r) * 4096) + k0;
      v16h ap;
#pragma unroll
      for (int j = 0; j < 8; ++j) {
        ap[j]     = prow[hi * 8 + j];
        ap[8 + j] = prow[16 + hi * 8 + j];
      }
      v16h bvr = load_bT_f16(vtrh, kS, dt * 16, k0, lane);
      v16h bvi = load_bT_f16(vtih, kS, dt * 16, k0, lane);
      accr = wmma32(ap, bvr, accr);
      acci = wmma32(ap, bvi, acci);
    }
#pragma unroll
    for (int j = 0; j < 8; ++j) {
      const int rowL = w * 16 + j + 8 * hi;
      const int row  = qrow0 + j + 8 * hi;
      const int col  = dt * 16 + r;
      const float sc = sinv[rowL];
      a_r[(size_t)split * kHSD + (size_t)row * kD + col] = accr[j] * sc;
      a_i[(size_t)split * kHSD + (size_t)row * kD + col] = acci[j] * sc;
    }
  }
}

// ---------------------------------------------------------------------------
// Combine: joint RMS over 2D concat, lambda-weighted difference, gating.
// ---------------------------------------------------------------------------
__global__ void cda_combine_kernel(const float* __restrict__ a_r,
                                   const float* __restrict__ a_i,
                                   const float* __restrict__ g_r,
                                   const float* __restrict__ g_i,
                                   const float* __restrict__ sub_w,
                                   const float* __restrict__ lq1,
                                   const float* __restrict__ lk1,
                                   const float* __restrict__ lq2,
                                   const float* __restrict__ lk2,
                                   float* __restrict__ y_r,
                                   float* __restrict__ y_i) {
  const int t = blockIdx.x;
  const int d = threadIdx.x;  // 0..127
  const size_t base = (size_t)t * kD + d;

  const float a1r = a_r[base];
  const float a1i = a_i[base];
  const float a2r = a_r[(size_t)kHSD + base];
  const float a2i = a_i[(size_t)kHSD + base];

  __shared__ float r0[128], r1[128], r2[128];
  r0[d] = a1r * a1r + a1i * a1i + a2r * a2r + a2i * a2i;
  r1[d] = lq1[d] * lk1[d];
  r2[d] = lq2[d] * lk2[d];
  __syncthreads();
  for (int s = 64; s > 0; s >>= 1) {
    if (d < s) {
      r0[d] += r0[d + s];
      r1[d] += r1[d + s];
      r2[d] += r2[d + s];
    }
    __syncthreads();
  }
  const float inv = 1.0f / sqrtf(r0[0] * (1.0f / 256.0f) + 1e-5f);
  const float lam =
      1.0f / (1.0f + __expf(-(__expf(r1[0]) - __expf(r2[0]) + kLambdaInit)));

  const float n1r = a1r * inv * sub_w[d];
  const float n1i = a1i * inv * sub_w[d];
  const float n2r = a2r * inv * sub_w[kD + d];
  const float n2i = a2i * inv * sub_w[kD + d];
  const float o_r = n1r - lam * n2r;
  const float o_i = n1i - lam * n2i;

  const float gr = g_r[base];
  const float gi = g_i[base];
  y_r[base] = gr * o_r - gi * o_i;
  y_i[base] = gr * o_i + gi * o_r;
}

// ---------------------------------------------------------------------------
extern "C" void kernel_launch(void* const* d_in, const int* in_sizes, int n_in,
                              void* d_out, int out_size, void* d_ws,
                              size_t ws_size, hipStream_t stream) {
  (void)in_sizes; (void)n_in; (void)out_size; (void)ws_size;

  const float* q_r    = (const float*)d_in[0];
  const float* q_i    = (const float*)d_in[1];
  const float* k_r    = (const float*)d_in[2];
  const float* k_i    = (const float*)d_in[3];
  const float* v_r    = (const float*)d_in[4];
  const float* v_i    = (const float*)d_in[5];
  const float* pe_q_r = (const float*)d_in[6];
  const float* pe_q_i = (const float*)d_in[7];
  const float* pe_k_r = (const float*)d_in[8];
  const float* pe_k_i = (const float*)d_in[9];
  const float* qw_r   = (const float*)d_in[10];
  const float* qw_i   = (const float*)d_in[11];
  const float* qb_r   = (const float*)d_in[12];
  const float* qb_i   = (const float*)d_in[13];
  const float* kw_r   = (const float*)d_in[14];
  const float* kw_i   = (const float*)d_in[15];
  const float* kb_r   = (const float*)d_in[16];
  const float* kb_i   = (const float*)d_in[17];
  const float* vw_r   = (const float*)d_in[18];
  const float* vw_i   = (const float*)d_in[19];
  const float* vb_r   = (const float*)d_in[20];
  const float* vb_i   = (const float*)d_in[21];
  const float* gw_r   = (const float*)d_in[22];
  const float* gw_i   = (const float*)d_in[23];
  const float* gb_r   = (const float*)d_in[24];
  const float* gb_i   = (const float*)d_in[25];
  const float* ow_r   = (const float*)d_in[26];
  const float* ow_i   = (const float*)d_in[27];
  const float* ob_r   = (const float*)d_in[28];
  const float* ob_i   = (const float*)d_in[29];
  const float* lq1    = (const float*)d_in[30];
  const float* lk1    = (const float*)d_in[31];
  const float* lq2    = (const float*)d_in[32];
  const float* lk2    = (const float*)d_in[33];
  const float* sub_w  = (const float*)d_in[34];

  // ---- workspace layout (~75.5 MB) ----
  char* ws = (char*)d_ws;
  size_t off = 0;
  auto carve = [&](size_t bytes) -> void* {
    void* p = ws + off;
    off += (bytes + 255) & ~(size_t)255;
    return p;
  };
  _Float16* qp16_r = (_Float16*)carve((size_t)kHS * 2 * kD * sizeof(_Float16));
  _Float16* qp16_i = (_Float16*)carve((size_t)kHS * 2 * kD * sizeof(_Float16));
  _Float16* kp16_r = (_Float16*)carve((size_t)kHS * kD * sizeof(_Float16));
  _Float16* kp16_i = (_Float16*)carve((size_t)kHS * kD * sizeof(_Float16));
  _Float16* vt16_r = (_Float16*)carve((size_t)kHS * kD * sizeof(_Float16));
  _Float16* vt16_i = (_Float16*)carve((size_t)kHS * kD * sizeof(_Float16));
  float*    g32_r  = (float*)carve((size_t)kHSD * sizeof(float));
  float*    g32_i  = (float*)carve((size_t)kHSD * sizeof(float));
  float*    a32_r  = (float*)carve((size_t)2 * kHSD * sizeof(float));
  float*    a32_i  = (float*)carve((size_t)2 * kHSD * sizeof(float));
  float*    y32_r  = (float*)carve((size_t)kHSD * sizeof(float));
  float*    y32_i  = (float*)carve((size_t)kHSD * sizeof(float));

  const dim3 blk32(32);
  const int  mTiles = kHS / 16;  // 768

  // q projection -> [HS, 2D] f16 (+pe duplicated)
  cda_proj_kernel<<<dim3(mTiles, (2 * kD) / 16), blk32, 0, stream>>>(
      q_r, q_i, qw_r, qw_i, qb_r, qb_i, pe_q_r, pe_q_i, 2 * kD, PROJ_ROW16,
      qp16_r, qp16_i, nullptr, nullptr);
  // k projection -> [HS, D] f16 (+pe_k)
  cda_proj_kernel<<<dim3(mTiles, kD / 16), blk32, 0, stream>>>(
      k_r, k_i, kw_r, kw_i, kb_r, kb_i, pe_k_r, pe_k_i, kD, PROJ_ROW16,
      kp16_r, kp16_i, nullptr, nullptr);
  // v projection -> transposed [H, D, S] f16
  cda_proj_kernel<<<dim3(mTiles, kD / 16), blk32, 0, stream>>>(
      v_r, v_i, vw_r, vw_i, vb_r, vb_i, nullptr, nullptr, kD, PROJ_TR16,
      vt16_r, vt16_i, nullptr, nullptr);
  // gate projection (from q!) -> [HS, D] f32
  cda_proj_kernel<<<dim3(mTiles, kD / 16), blk32, 0, stream>>>(
      q_r, q_i, gw_r, gw_i, gb_r, gb_i, nullptr, nullptr, kD, PROJ_F32,
      nullptr, nullptr, g32_r, g32_i);

  // attention (both splits): 32 q-blocks x 12 heads x 2 splits
  cda_attn_kernel<<<dim3(kS / 32, kH, 2), dim3(64), kAttnLds, stream>>>(
      qp16_r, qp16_i, kp16_r, kp16_i, vt16_r, vt16_i, a32_r, a32_i);

  // rms-norm + lambda difference + gating
  cda_combine_kernel<<<dim3(kHS), dim3(128), 0, stream>>>(
      a32_r, a32_i, g32_r, g32_i, sub_w, lq1, lk1, lq2, lk2, y32_r, y32_i);

  // output projection -> d_out (out_r then out_i, each [HS, D] f32)
  cda_proj_kernel<<<dim3(mTiles, kD / 16), blk32, 0, stream>>>(
      y32_r, y32_i, ow_r, ow_i, ob_r, ob_i, nullptr, nullptr, kD, PROJ_F32,
      nullptr, nullptr, (float*)d_out, (float*)d_out + kHSD);
}
